// StatefulCausalAttentionHead_12558484374034
// MI455X (gfx1250) — compile-verified, matrix operand order checked by reading, not run on hardware
//
#include <hip/hip_runtime.h>
#include <hip/hip_bf16.h>

// ---------------------------------------------------------------------------
// StatefulCausalAttentionHead for MI455X (gfx1250, wave32, WMMA + async/TDM)
//   B=4, SEQ=4096, STATE=128, T=4352, DIN=512, DK=DV=64, SCALE=0.125
// bf16 WMMA (f32 accumulate) for QKV projection, QK^T and PV.
// Attention: 4-wave workgroups, K/V tiles double-buffered in LDS via
// global_load_async_to_lds_b128 (ASYNCcnt) and tensor_load_to_lds (TDM,
// TENSORcnt), overlapping staging with WMMA compute.
// ---------------------------------------------------------------------------

#define B_      4
#define SEQ_    4096
#define STATE_  128
#define T_      (SEQ_ + 2 * STATE_)     // 4352
#define DIN_    512
#define DK_     64
#define TILES_  (T_ / 16)               // 272
#define KSTEPS_ (DIN_ / 32)             // 16
#define SCALE_  0.125f
#define QB_     64                      // q rows per workgroup (4 waves x 16)
#define KT_     32                      // key-tile size

typedef __attribute__((ext_vector_type(16))) __bf16 bf16x16;  // 8 VGPRs
typedef __attribute__((ext_vector_type(8)))  float  f32x8;    // 8 VGPRs
typedef unsigned int u32x4 __attribute__((ext_vector_type(4)));
typedef int          i32x4 __attribute__((ext_vector_type(4)));
typedef int          i32x8 __attribute__((ext_vector_type(8)));
typedef int          v4i   __attribute__((vector_size(16)));   // builtin ptr type

#define AS1_ __attribute__((address_space(1)))
#define AS3_ __attribute__((address_space(3)))

#if __has_builtin(__builtin_amdgcn_tensor_load_to_lds)
#define HAS_TDM 1
#else
#define HAS_TDM 0
#endif

static __device__ __forceinline__ f32x8 wmma_bf16(bf16x16 a, bf16x16 b, f32x8 c) {
  return __builtin_amdgcn_wmma_f32_16x16x32_bf16(false, a, false, b, (short)0, c,
                                                 false, false);
}

// Load a 16-bit A/B fragment: two contiguous 16-byte (8 x bf16) chunks.
static __device__ __forceinline__ bf16x16 ldfrag(const __bf16* c0, const __bf16* c1) {
  union { bf16x16 v; uint4 u[2]; } U;
  U.u[0] = *reinterpret_cast<const uint4*>(c0);
  U.u[1] = *reinterpret_cast<const uint4*>(c1);
  return U.v;
}

// Flat LDS addresses keep the LDS byte offset in the low 32 bits (ISA 10.2).
static __device__ __forceinline__ unsigned lds32(const void* p) {
  return (unsigned)(unsigned long long)p;
}

// Async 16B global->LDS copy (tracked by ASYNCcnt).
static __device__ __forceinline__ void async_copy16(void* lds, const void* g) {
#if __has_builtin(__builtin_amdgcn_global_load_async_to_lds_b128)
  __builtin_amdgcn_global_load_async_to_lds_b128(
      (AS1_ v4i*)(unsigned long long)g,
      (AS3_ v4i*)lds32(lds), 0, 0);
#else
  asm volatile("global_load_async_to_lds_b128 %0, %1, off"
               :: "v"(lds32(lds)), "v"((unsigned long long)g)
               : "memory");
#endif
}

template <int N>
static __device__ __forceinline__ void wait_asynccnt() {
#if __has_builtin(__builtin_amdgcn_s_wait_asynccnt)
  __builtin_amdgcn_s_wait_asynccnt(N);
#else
  asm volatile("s_wait_asynccnt %0" :: "i"(N) : "memory");
#endif
}

template <int N>
static __device__ __forceinline__ void wait_tensorcnt() {
#if __has_builtin(__builtin_amdgcn_s_wait_tensorcnt)
  __builtin_amdgcn_s_wait_tensorcnt(N);
#else
  asm volatile("s_wait_tensorcnt %0" :: "i"(N) : "memory");
#endif
}

#if HAS_TDM
// TDM: stage a 2-D tile (tile_dim0=KT_ keys along dim0, tile_dim1=DK_ rows)
// of the V^T tensor (tensor_dim0=T_, stride T_, data_size=2B) into LDS.
// D# packing per CDNA5 ISA sec. 8.3/8.4 (group0: count/lds/global/type,
// group1: data_size, dims, tile dims, strides; groups 2/3 zero = <=2D).
static __device__ __forceinline__ void tdm_stage_v(unsigned ldsAddr,
                                                   const void* gaddr) {
  unsigned long long ga = (unsigned long long)gaddr;
  u32x4 g0 = { 1u,                                   // count=1, user mode
               ldsAddr,                              // lds_addr
               (unsigned)ga,                         // global_addr[31:0]
               (unsigned)((ga >> 32) & 0x1FFFFFFull) | (2u << 30) }; // type=2
  i32x8 g1 = { 0x00010000,                           // wg_mask=0, data_size=2B
               (int)((T_ & 0xFFFF) << 16),           // tensor_dim0 lo16
               (int)(((T_ >> 16) & 0xFFFF) | ((DK_ & 0xFFFF) << 16)), // d0hi|d1lo
               (int)((unsigned)KT_ << 16),           // d1 hi | tile_dim0=KT_
               (int)DK_,                             // tile_dim1=DK_, tile_dim2=0
               (int)T_,                              // tensor_dim0_stride lo32
               0, 0 };
  i32x4 g2 = { 0, 0, 0, 0 };
  i32x4 g3 = { 0, 0, 0, 0 };
#if __clang_major__ >= 23
  i32x8 g1b = { 0, 0, 0, 0, 0, 0, 0, 0 };
  __builtin_amdgcn_tensor_load_to_lds(g0, g1, g2, g3, g1b, 0);
#else
  __builtin_amdgcn_tensor_load_to_lds(g0, g1, g2, g3, 0);
#endif
}
#endif

// ---------------------------------------------------------------------------
// Kernel 1: swizzle the 6 fp32 512x64 weight matrices into bf16 WMMA
// B-fragment order: [widx][ks][nt][lane][e], 16 bf16 (32B) contiguous/lane.
// ---------------------------------------------------------------------------
__global__ void prep_weights_kernel(const float* w0, const float* w1,
                                    const float* w2, const float* w3,
                                    const float* w4, const float* w5,
                                    __bf16* out) {
  int idx = blockIdx.x * blockDim.x + threadIdx.x;
  const int per_mat = KSTEPS_ * 4 * 32 * 16;   // 32768
  if (idx >= 6 * per_mat) return;
  int e    = idx & 15;
  int lane = (idx >> 4) & 31;
  int nt   = (idx >> 9) & 3;
  int ks   = (idx >> 11) & (KSTEPS_ - 1);
  int wsel = idx >> 15;
  const float* W = (wsel == 0) ? w0 : (wsel == 1) ? w1 : (wsel == 2) ? w2
                 : (wsel == 3) ? w3 : (wsel == 4) ? w4 : w5;
  int n = lane & 15, h = lane >> 4;
  int kk = (e < 8) ? (8 * h + e) : (8 * h + e + 8);
  int k  = ks * 32 + kk;
  out[idx] = (__bf16)W[k * DK_ + nt * 16 + n];
}

// ---------------------------------------------------------------------------
// Kernel 2: QKV projection. One wave computes a 16x64 tile = x(16x512)@W.
// Q,K row-major bf16; V transposed (64 x T) bf16 so PV B-frags are contiguous.
// ---------------------------------------------------------------------------
__global__ void qkv_proj_kernel(const float* __restrict__ x,
                                const __bf16* __restrict__ wsw,
                                __bf16* __restrict__ Qb,
                                __bf16* __restrict__ Kb,
                                __bf16* __restrict__ Vt) {
  const int tile = blockIdx.x;
  const int sel  = blockIdx.y;                 // 0=q 1=k 2=v
  const int b    = tile / TILES_;
  const int tl   = tile % TILES_;
  const bool state = (tl < STATE_ / 16) || (tl >= STATE_ / 16 + SEQ_ / 16);
  const int widx = sel + (state ? 3 : 0);
  const __bf16* wbase = wsw + (size_t)widx * (KSTEPS_ * 4 * 32 * 16);

  const int lane = threadIdx.x;
  const int n = lane & 15, h = lane >> 4;
  const float* xrow = x + ((size_t)b * T_ + tl * 16 + n) * DIN_;

  f32x8 acc[4] = {};
  for (int ks = 0; ks < KSTEPS_; ++ks) {
    const int k0 = ks * 32 + 8 * h;
    float4 c0 = *reinterpret_cast<const float4*>(xrow + k0);
    float4 c1 = *reinterpret_cast<const float4*>(xrow + k0 + 4);
    float4 c2 = *reinterpret_cast<const float4*>(xrow + k0 + 16);
    float4 c3 = *reinterpret_cast<const float4*>(xrow + k0 + 20);
    bf16x16 a;
    a[0] = (__bf16)c0.x;  a[1] = (__bf16)c0.y;  a[2] = (__bf16)c0.z;  a[3] = (__bf16)c0.w;
    a[4] = (__bf16)c1.x;  a[5] = (__bf16)c1.y;  a[6] = (__bf16)c1.z;  a[7] = (__bf16)c1.w;
    a[8] = (__bf16)c2.x;  a[9] = (__bf16)c2.y;  a[10] = (__bf16)c2.z; a[11] = (__bf16)c2.w;
    a[12] = (__bf16)c3.x; a[13] = (__bf16)c3.y; a[14] = (__bf16)c3.z; a[15] = (__bf16)c3.w;

    const __bf16* bks = wbase + (size_t)(ks * 4) * 32 * 16;
#pragma unroll
    for (int nt = 0; nt < 4; ++nt) {
      const __bf16* bl = bks + (size_t)(nt * 32 + lane) * 16;
      bf16x16 bf = ldfrag(bl, bl + 8);
      acc[nt] = wmma_bf16(a, bf, acc[nt]);
    }
  }

  if (sel == 2) {
#pragma unroll
    for (int nt = 0; nt < 4; ++nt) {
      const int f = nt * 16 + n;
      __bf16* dst = Vt + ((size_t)b * DK_ + f) * T_ + tl * 16 + 8 * h;
#pragma unroll
      for (int r = 0; r < 8; ++r) dst[r] = (__bf16)acc[nt][r];
    }
  } else {
    __bf16* dst = (sel == 0 ? Qb : Kb) + ((size_t)b * T_ + tl * 16) * DK_;
#pragma unroll
    for (int nt = 0; nt < 4; ++nt)
#pragma unroll
      for (int r = 0; r < 8; ++r)
        dst[(size_t)(r + 8 * h) * DK_ + nt * 16 + n] = (__bf16)acc[nt][r];
  }
}

// ---------------------------------------------------------------------------
// Kernel 3: causal flash attention. 4 waves/block, 16 q rows each; K/V tiles
// staged cooperatively into double-buffered LDS: K via async-LDS loads
// (ASYNCcnt), V via TDM descriptor (TENSORcnt) when available. In-order
// completion lets "wait <= in-flight-of-next-tile" overlap stage & compute.
// ---------------------------------------------------------------------------
__global__ __launch_bounds__(128) void attn_kernel(const __bf16* __restrict__ Qb,
                                                   const __bf16* __restrict__ Kb,
                                                   const __bf16* __restrict__ Vt,
                                                   float* __restrict__ out) {
  __shared__ __align__(16) __bf16 kbuf[2][KT_ * DK_];   // 32 keys x 64d (4KB)
  __shared__ __align__(16) __bf16 vbuf[2][DK_ * KT_];   // 64f x 32 keys (4KB)
  __shared__ __align__(16) __bf16 pbuf[4][16 * 32];     // per-wave P tiles

  const int b    = blockIdx.x / (T_ / QB_);
  const int qb   = blockIdx.x % (T_ / QB_);
  const int tid  = threadIdx.x;
  const int wave = tid >> 5;
  const int lane = tid & 31;
  const int n = lane & 15, h = lane >> 4;
  const int qbase = qb * QB_ + wave * 16;

  const __bf16* kgbase = Kb + (size_t)b * T_ * DK_;
  const __bf16* vgbase = Vt + (size_t)b * DK_ * T_;

  // K tile is 32 contiguous 128B rows -> 4KB contiguous: 2 async b128/thread.
  auto stage_k = [&](int buf, int kbase) {
    const char* src = (const char*)(kgbase + (size_t)kbase * DK_);
    char* dst = (char*)&kbuf[buf][0];
    async_copy16(dst + tid * 16,        src + tid * 16);
    async_copy16(dst + tid * 16 + 2048, src + tid * 16 + 2048);
  };
#if HAS_TDM
  constexpr int APT = 2;   // async ops/thread per stage (K only)
  auto stage_v = [&](int buf, int kbase) {
    if (wave == 0) tdm_stage_v(lds32(&vbuf[buf][0]), vgbase + kbase);
  };
#else
  constexpr int APT = 4;   // K (2) + V (2)
  auto stage_v = [&](int buf, int kbase) {
    const char* src = (const char*)(vgbase + kbase);
    char* dst = (char*)&vbuf[buf][0];
    const int c0 = tid, c1 = tid + 128;   // 64 rows x 4 chunks of 16B
    async_copy16(dst + c0 * 16,
                 src + (size_t)(c0 >> 2) * (T_ * 2) + (c0 & 3) * 16);
    async_copy16(dst + c1 * 16,
                 src + (size_t)(c1 >> 2) * (T_ * 2) + (c1 & 3) * 16);
  };
#endif

  // Q A-fragments held in registers for the whole loop.
  const __bf16* qrow = Qb + ((size_t)b * T_ + qbase + n) * DK_;
  const bf16x16 a0 = ldfrag(qrow + 8 * h,      qrow + 16 + 8 * h);
  const bf16x16 a1 = ldfrag(qrow + 32 + 8 * h, qrow + 48 + 8 * h);

  float mrow[8], lrow[8];
  f32x8 acc[4] = {};
#pragma unroll
  for (int r = 0; r < 8; ++r) { mrow[r] = -INFINITY; lrow[r] = 0.0f; }

  const int nkt = (qb * QB_ + QB_ + KT_ - 1) / KT_;   // block-diagonal bound

  stage_k(0, 0);
  stage_v(0, 0);

  for (int kt = 0; kt < nkt; ++kt) {
    const int buf = kt & 1;
    const int kbase = kt * KT_;
    const bool more = (kt + 1 < nkt);

    if (more) {                 // prefetch next tile into the other buffer
      stage_k(buf ^ 1, kbase + KT_);
      stage_v(buf ^ 1, kbase + KT_);
      wait_asynccnt<APT>();     // in-order: current tile's K loads are done
    } else {
      wait_asynccnt<0>();
    }
#if HAS_TDM
    if (wave == 0) { if (more) wait_tensorcnt<1>(); else wait_tensorcnt<0>(); }
#endif
    __syncthreads();            // staged data visible to all waves

    if (kbase <= qbase + 15) {  // this wave still has keys in this tile
      // --- S = Q K^T from LDS K tile -------------------------------------
      const __bf16* kb0 = &kbuf[buf][(size_t)n * DK_];
      const __bf16* kb1 = &kbuf[buf][(size_t)(16 + n) * DK_];
      bf16x16 kb00 = ldfrag(kb0 + 8 * h,      kb0 + 16 + 8 * h);
      bf16x16 kb01 = ldfrag(kb0 + 32 + 8 * h, kb0 + 48 + 8 * h);
      bf16x16 kb10 = ldfrag(kb1 + 8 * h,      kb1 + 16 + 8 * h);
      bf16x16 kb11 = ldfrag(kb1 + 32 + 8 * h, kb1 + 48 + 8 * h);

      f32x8 s0 = {}, s1 = {};
      s0 = wmma_bf16(a0, kb00, s0);
      s0 = wmma_bf16(a1, kb01, s0);
      s1 = wmma_bf16(a0, kb10, s1);
      s1 = wmma_bf16(a1, kb11, s1);

      const bool diag = (kbase + KT_ - 1 > qbase);

      // --- online softmax (rows live in 16-lane C-fragment halves) --------
#pragma unroll
      for (int r = 0; r < 8; ++r) {
        const int row = qbase + r + 8 * h;
        float v0 = s0[r] * SCALE_;
        float v1 = s1[r] * SCALE_;
        if (diag) {
          if (kbase + n > row)      v0 = -INFINITY;
          if (kbase + 16 + n > row) v1 = -INFINITY;
        }
        float rm = fmaxf(v0, v1);
        rm = fmaxf(rm, __shfl_xor(rm, 1));
        rm = fmaxf(rm, __shfl_xor(rm, 2));
        rm = fmaxf(rm, __shfl_xor(rm, 4));
        rm = fmaxf(rm, __shfl_xor(rm, 8));
        const float mnew = fmaxf(mrow[r], rm);
        const float p0 = __expf(v0 - mnew);
        const float p1 = __expf(v1 - mnew);
        float rs = p0 + p1;
        rs += __shfl_xor(rs, 1);
        rs += __shfl_xor(rs, 2);
        rs += __shfl_xor(rs, 4);
        rs += __shfl_xor(rs, 8);
        const float alpha = __expf(mrow[r] - mnew);
        lrow[r] = lrow[r] * alpha + rs;
        mrow[r] = mnew;
#pragma unroll
        for (int nt = 0; nt < 4; ++nt) acc[nt][r] *= alpha;
        pbuf[wave][(r + 8 * h) * 32 + n]      = (__bf16)p0;
        pbuf[wave][(r + 8 * h) * 32 + 16 + n] = (__bf16)p1;
      }
      // Per-wave LDS is in-order (DScnt): no barrier needed for the repack.
      const __bf16* prow = &pbuf[wave][(size_t)n * 32];
      bf16x16 pf = ldfrag(prow + 8 * h, prow + 16 + 8 * h);

      // --- O += P V from LDS V^T tile ------------------------------------
#pragma unroll
      for (int nt = 0; nt < 4; ++nt) {
        const __bf16* vp = &vbuf[buf][(size_t)(nt * 16 + n) * KT_];
        bf16x16 vf = ldfrag(vp + 8 * h, vp + 16 + 8 * h);
        acc[nt] = wmma_bf16(pf, vf, acc[nt]);
      }
    }
    __syncthreads();            // all reads of buf done before it is restaged
  }

  // --- epilogue: out = acc / l --------------------------------------------
  float* orow = out + ((size_t)b * T_ + qbase) * DK_;
#pragma unroll
  for (int nt = 0; nt < 4; ++nt)
#pragma unroll
    for (int r = 0; r < 8; ++r)
      orow[(size_t)(r + 8 * h) * DK_ + nt * 16 + n] = acc[nt][r] / lrow[r];
}

// ---------------------------------------------------------------------------
// Launcher. Inputs: x, Wq, Wk, Wv, Wq_s, Wk_s, Wv_s (all fp32).
// Workspace: swizzled bf16 weights (384KB) | Q bf16 | K bf16 | V^T bf16
// (~6.75 MB total).
// ---------------------------------------------------------------------------
extern "C" void kernel_launch(void* const* d_in, const int* in_sizes, int n_in,
                              void* d_out, int out_size, void* d_ws, size_t ws_size,
                              hipStream_t stream) {
  const float* x  = (const float*)d_in[0];
  const float* w0 = (const float*)d_in[1];
  const float* w1 = (const float*)d_in[2];
  const float* w2 = (const float*)d_in[3];
  const float* w3 = (const float*)d_in[4];
  const float* w4 = (const float*)d_in[5];
  const float* w5 = (const float*)d_in[6];

  const size_t wswBytes = (size_t)6 * KSTEPS_ * 4 * 32 * 16 * sizeof(__bf16);
  const size_t qkvBytes = (size_t)B_ * T_ * DK_ * sizeof(__bf16);
  char* base = (char*)d_ws;
  __bf16* wsw = (__bf16*)base;
  __bf16* Qb  = (__bf16*)(base + wswBytes);
  __bf16* Kb  = (__bf16*)(base + wswBytes + qkvBytes);
  __bf16* Vt  = (__bf16*)(base + wswBytes + 2 * qkvBytes);

  const int wtot = 6 * KSTEPS_ * 4 * 32 * 16;
  prep_weights_kernel<<<(wtot + 255) / 256, 256, 0, stream>>>(w0, w1, w2, w3,
                                                              w4, w5, wsw);
  qkv_proj_kernel<<<dim3(B_ * TILES_, 3), 32, 0, stream>>>(x, wsw, Qb, Kb, Vt);
  attn_kernel<<<B_ * (T_ / QB_), 128, 0, stream>>>(Qb, Kb, Vt, (float*)d_out);
}
